// DualHierarchicalGNNUNet_15985868275833
// MI455X (gfx1250) — matrix operand order repaired and checked by reading, not compiled, not run on hardware
//
#include <hip/hip_runtime.h>
#include <hip/hip_bf16.h>

// ---------------------------------------------------------------------------
// CDNA5 (gfx1250) implementation: bf16 WMMA everywhere a GEMM shape exists,
// async global->LDS staging (ASYNCcnt) for the attention V tiles.
// ---------------------------------------------------------------------------

typedef __attribute__((ext_vector_type(16))) __bf16 v16bf;
typedef __attribute__((ext_vector_type(8)))  float  v8f;

union V16 { v16bf v; uint4 q[2]; unsigned short s[16]; };

static constexpr int Bn    = 2;
static constexpr int E     = 128;
static constexpr int HW    = 4096;   // 64*64
static constexpr int NHEAD = 4;
static constexpr int FCn   = 128;
static constexpr int CP1   = 288;    // 257 padded to multiple of 32
static constexpr int CP2   = 128;
static constexpr int MTOT  = Bn * HW; // 8192

__device__ __forceinline__ unsigned short f2bf(float f) {
    union { float f; unsigned u; } x; x.f = f;
    unsigned r = x.u + 0x7FFFu + ((x.u >> 16) & 1u);
    return (unsigned short)(r >> 16);
}
__device__ __forceinline__ float bf2f(unsigned short s) {
    union { unsigned u; float f; } x; x.u = ((unsigned)s) << 16;
    return x.f;
}
__device__ __forceinline__ v8f wmma_bf16(v16bf a, v16bf b, v8f c) {
    return __builtin_amdgcn_wmma_f32_16x16x32_bf16(false, a, false, b, (short)0, c,
                                                   false, false);
}

// ---- layout converts ------------------------------------------------------

// NCHW f32 -> (B*HW, C) bf16 channels-last
__global__ void k_nchw_nlc(const float* __restrict__ src, unsigned short* __restrict__ dst,
                           int C, int HWn, int total) {
    int i = blockIdx.x * blockDim.x + threadIdx.x;
    if (i >= total) return;
    int b = i / (HWn * C);
    int rem = i - b * HWn * C;
    int p = rem / C, c = rem - p * C;
    dst[i] = f2bf(src[(b * C + c) * HWn + p]);
}

__global__ void k_cvt(const float* __restrict__ s, unsigned short* __restrict__ d, int n) {
    int i = blockIdx.x * blockDim.x + threadIdx.x;
    if (i < n) d[i] = f2bf(s[i]);
}

// (FC, Cin, 3, 3) f32 -> (FC, 9, Cpad) bf16, zero-padded channels
__global__ void k_convw(const float* __restrict__ w, unsigned short* __restrict__ d,
                        int Cin, int Cpad) {
    int i = blockIdx.x * blockDim.x + threadIdx.x;
    int total = FCn * 9 * Cpad;
    if (i >= total) return;
    int fc = i / (9 * Cpad);
    int rem = i - fc * 9 * Cpad;
    int r = rem / Cpad, c = rem - r * Cpad;
    d[i] = (c < Cin) ? f2bf(w[(fc * Cin + c) * 9 + r]) : (unsigned short)0;
}

// gnn channel + zero pad into fused[..., 256..287]
__global__ void k_fused_tail(const float* __restrict__ gnn, unsigned short* __restrict__ fused) {
    int i = blockIdx.x * blockDim.x + threadIdx.x;
    if (i >= MTOT * 32) return;
    int p = i >> 5, c = i & 31;
    fused[p * CP1 + 256 + c] = (c == 0) ? f2bf(gnn[p]) : (unsigned short)0;
}

// ---- generic WMMA GEMM: C = A(MxK bf16) * Bw(NxK bf16)^T + bias -----------

__global__ void k_gemm(const unsigned short* __restrict__ A, int lda,
                       const unsigned short* __restrict__ Bw, int ldb,
                       const float* __restrict__ bias,
                       unsigned short* __restrict__ Cb, float* __restrict__ Cf,
                       int ldc, int colOff, int Mtiles, int Ntiles, int Ksteps, int relu) {
    int wid  = (blockIdx.x * blockDim.x + threadIdx.x) >> 5;
    int lane = threadIdx.x & 31;
    if (wid >= Mtiles * Ntiles) return;
    int mt = wid / Ntiles, nt = wid - mt * Ntiles;
    int g = lane >> 4, nn = lane & 15;

    const unsigned short* Arow = A + (size_t)(mt * 16 + nn) * lda;
    const unsigned short* Brow = Bw + (size_t)(nt * 16 + nn) * ldb;
    v8f c = {};
    for (int ks = 0; ks < Ksteps; ++ks) {
        int kb = ks * 32;
        V16 a, b;
        a.q[0] = *reinterpret_cast<const uint4*>(Arow + kb + 8 * g);
        a.q[1] = *reinterpret_cast<const uint4*>(Arow + kb + 16 + 8 * g);
        b.q[0] = *reinterpret_cast<const uint4*>(Brow + kb + 8 * g);
        b.q[1] = *reinterpret_cast<const uint4*>(Brow + kb + 16 + 8 * g);
        c = wmma_bf16(a.v, b.v, c);
    }
    int ncol = nt * 16 + nn;
    float bia = bias ? bias[ncol] : 0.f;
    for (int r = 0; r < 8; ++r) {
        int m = mt * 16 + r + 8 * g;
        float v = c[r] + bia;
        if (relu) v = fmaxf(v, 0.f);
        if (Cb) Cb[(size_t)m * ldc + colOff + ncol] = f2bf(v);
        else    Cf[(size_t)m * ldc + colOff + ncol] = v;
    }
}

// ---- flash attention: one wave per 16-query tile per (batch, head) --------
// V chunks (32 keys x 32 dims bf16 = 2KB) are streamed into LDS with
// global_load_async_to_lds_b128, double-buffered (ASYNCcnt pipeline).

__global__ void k_attn(const unsigned short* __restrict__ Q,
                       const unsigned short* __restrict__ K,
                       const unsigned short* __restrict__ V,
                       unsigned short* __restrict__ Ctx) {
    int tile = blockIdx.x;                // B*NHEAD*(HW/16) tiles
    int qt = tile & 255;
    int hb = tile >> 8;
    int h  = hb & (NHEAD - 1);
    int b  = hb >> 2;
    int q0 = b * HW + qt * 16;
    int hd = h * 32;
    int lane = threadIdx.x;
    int g = lane >> 4, nn = lane & 15;

    __shared__ unsigned short ldsP[16 * 32];
    __shared__ unsigned short ldsV[2][32 * 32];

    unsigned long long vbase = (unsigned long long)(size_t)V;
    // Issue one 32x32 bf16 V chunk into LDS buffer `buf` (4 x 512B async copies)
    auto issueV = [&](int kc, int buf) {
        for (int i = 0; i < 4; ++i) {
            unsigned oo  = (unsigned)(i * 512 + lane * 16);       // byte off in chunk
            unsigned row = oo >> 6, col = oo & 63;                // key row, byte col
            unsigned goff = (unsigned)((b * HW + kc * 32 + row) * (E * 2) + hd * 2 + col);
            unsigned laddr = (unsigned)(size_t)(&ldsV[buf][0]) + oo;
            asm volatile("global_load_async_to_lds_b128 %0, %1, %2"
                         :: "v"(laddr), "v"(goff), "s"(vbase) : "memory");
        }
    };

    const unsigned short* Qrow = Q + (size_t)(q0 + nn) * E + hd;
    V16 aq;
    aq.q[0] = *reinterpret_cast<const uint4*>(Qrow + 8 * g);
    aq.q[1] = *reinterpret_cast<const uint4*>(Qrow + 16 + 8 * g);

    float mrow[8], lsum[8];
    v8f o0 = {}, o1 = {};
    for (int r = 0; r < 8; ++r) { mrow[r] = -1e30f; lsum[r] = 0.f; }
    const float sc = 0.17677669529663687f;   // 1/sqrt(32)

    issueV(0, 0);

    for (int kc = 0; kc < 128; ++kc) {
        int cur = kc & 1;
        int k0 = b * HW + kc * 32;
        if (kc + 1 < 128)  // pull next K chunk toward L2/L0 (global_prefetch_b8)
            __builtin_prefetch(K + (size_t)(k0 + 32 + nn) * E + hd, 0, 1);

        // scores: two 16x16 tiles (K-dim = head dim = 32)
        const unsigned short* Kp0 = K + (size_t)(k0 + nn) * E + hd + 8 * g;
        const unsigned short* Kp1 = K + (size_t)(k0 + 16 + nn) * E + hd + 8 * g;
        V16 b0, b1;
        b0.q[0] = *reinterpret_cast<const uint4*>(Kp0);
        b0.q[1] = *reinterpret_cast<const uint4*>(Kp0 + 16);
        b1.q[0] = *reinterpret_cast<const uint4*>(Kp1);
        b1.q[1] = *reinterpret_cast<const uint4*>(Kp1 + 16);
        v8f s0 = {}, s1 = {};
        s0 = wmma_bf16(aq.v, b0.v, s0);
        s1 = wmma_bf16(aq.v, b1.v, s1);

        // online softmax per row (row m = r + 8*g lives across 16 lanes)
        for (int r = 0; r < 8; ++r) {
            float x0 = s0[r] * sc, x1 = s1[r] * sc;
            float v = fmaxf(x0, x1);
            for (int m = 8; m >= 1; m >>= 1) v = fmaxf(v, __shfl_xor(v, m));
            float mnew = fmaxf(mrow[r], v);
            float corr = __expf(mrow[r] - mnew);
            float p0 = __expf(x0 - mnew), p1 = __expf(x1 - mnew);
            float rs = p0 + p1;
            for (int m = 8; m >= 1; m >>= 1) rs += __shfl_xor(rs, m);
            lsum[r] = lsum[r] * corr + rs;
            mrow[r] = mnew;
            o0[r] *= corr; o1[r] *= corr;
            int mq = r + 8 * g;
            ldsP[mq * 32 + nn]      = f2bf(p0);
            ldsP[mq * 32 + 16 + nn] = f2bf(p1);
        }

        // overlap: launch next V chunk copy, then wait for current chunk only
        if (kc + 1 < 128) {
            issueV(kc + 1, cur ^ 1);
            asm volatile("s_wait_asynccnt 4" ::: "memory");
        } else {
            asm volatile("s_wait_asynccnt 0" ::: "memory");
        }
        asm volatile("s_wait_dscnt 0" ::: "memory");

        // re-layout P (C-layout -> A-layout) through wave-private LDS
        V16 ap;
        for (int t = 0; t < 16; ++t)
            ap.s[t] = ldsP[nn * 32 + 16 * g + t];

        // V tiles from LDS: B element t -> key = 16*g + t, col = nn (+16)
        V16 bv0, bv1;
        for (int t = 0; t < 16; ++t) {
            bv0.s[t] = ldsV[cur][(16 * g + t) * 32 + nn];
            bv1.s[t] = ldsV[cur][(16 * g + t) * 32 + 16 + nn];
        }
        o0 = wmma_bf16(ap.v, bv0.v, o0);
        o1 = wmma_bf16(ap.v, bv1.v, o1);
    }
    for (int r = 0; r < 8; ++r) {
        float inv = 1.f / lsum[r];
        int row = q0 + r + 8 * g;
        Ctx[(size_t)row * E + hd + nn]      = f2bf(o0[r] * inv);
        Ctx[(size_t)row * E + hd + 16 + nn] = f2bf(o1[r] * inv);
    }
}

// ---- implicit-GEMM 3x3 conv (channels-last bf16 in, fp32 pre-BN out) ------

__global__ void k_conv(const unsigned short* __restrict__ X,
                       const unsigned short* __restrict__ Wc,
                       float* __restrict__ Out, int Cpad) {
    int wid  = (blockIdx.x * blockDim.x + threadIdx.x) >> 5;
    int lane = threadIdx.x & 31;
    int Ntiles = FCn / 16;                       // 8
    if (wid >= (MTOT / 16) * Ntiles) return;
    int mt = wid / Ntiles, nt = wid - mt * Ntiles;
    int g = lane >> 4, nn = lane & 15;

    int p = mt * 16 + nn;
    int b = p >> 12, pos = p & 4095;
    int y = pos >> 6, x = pos & 63;
    const unsigned short* Brow = Wc + (size_t)(nt * 16 + nn) * (9 * Cpad);
    uint4 z4; z4.x = z4.y = z4.z = z4.w = 0u;

    v8f c = {};
    for (int r = 0; r < 9; ++r) {
        int yy = y + r / 3 - 1, xx = x + r % 3 - 1;
        bool valid = ((unsigned)yy < 64u) && ((unsigned)xx < 64u);
        const unsigned short* Arow = X + (size_t)((b << 12) + yy * 64 + xx) * Cpad;
        const unsigned short* Bk = Brow + r * Cpad;
        for (int cc = 0; cc < Cpad / 32; ++cc) {
            int kb = cc * 32;
            V16 a, bw;
            if (valid) {
                a.q[0] = *reinterpret_cast<const uint4*>(Arow + kb + 8 * g);
                a.q[1] = *reinterpret_cast<const uint4*>(Arow + kb + 16 + 8 * g);
            } else { a.q[0] = z4; a.q[1] = z4; }
            bw.q[0] = *reinterpret_cast<const uint4*>(Bk + kb + 8 * g);
            bw.q[1] = *reinterpret_cast<const uint4*>(Bk + kb + 16 + 8 * g);
            c = wmma_bf16(a.v, bw.v, c);
        }
    }
    for (int rr = 0; rr < 8; ++rr) {
        int mrow = mt * 16 + rr + 8 * g;
        Out[(size_t)mrow * FCn + nt * 16 + nn] = c[rr];
    }
}

// ---- training-mode batchnorm (population stats) ---------------------------

__global__ void k_bnstats(const float* __restrict__ X, const float* __restrict__ gamma,
                          const float* __restrict__ beta,
                          float* __restrict__ scale, float* __restrict__ shift) {
    int c = blockIdx.x, tid = threadIdx.x;
    __shared__ float s1[256], s2[256];
    float a = 0.f, b2 = 0.f;
    for (int i = tid; i < MTOT; i += 256) {
        float v = X[(size_t)i * FCn + c];
        a += v; b2 += v * v;
    }
    s1[tid] = a; s2[tid] = b2; __syncthreads();
    for (int s = 128; s > 0; s >>= 1) {
        if (tid < s) { s1[tid] += s1[tid + s]; s2[tid] += s2[tid + s]; }
        __syncthreads();
    }
    if (tid == 0) {
        float mean = s1[0] / (float)MTOT;
        float var  = s2[0] / (float)MTOT - mean * mean;
        float sc = gamma[c] * rsqrtf(var + 1e-5f);
        scale[c] = sc;
        shift[c] = beta[c] - mean * sc;
    }
}

__global__ void k_bnapply(const float* __restrict__ X, const float* __restrict__ scale,
                          const float* __restrict__ shift, unsigned short* __restrict__ Y) {
    int i = blockIdx.x * blockDim.x + threadIdx.x;
    if (i >= MTOT * FCn) return;
    int c = i & (FCn - 1);
    float v = fmaxf(X[i] * scale[c] + shift[c], 0.f);
    Y[i] = f2bf(v);
}

// ---- 1-channel heads: pixel logits + MIL scores ---------------------------

__global__ void k_heads(const unsigned short* __restrict__ feat,
                        const unsigned short* __restrict__ abuf,
                        const float* __restrict__ ow, const float* __restrict__ ob,
                        const float* __restrict__ aw, const float* __restrict__ ab,
                        float* __restrict__ pl, float* __restrict__ sco) {
    int p = blockIdx.x * blockDim.x + threadIdx.x;
    if (p >= MTOT) return;
    float s1 = 0.f, s2 = 0.f;
    for (int c = 0; c < FCn; ++c) {
        s1 += bf2f(feat[(size_t)p * FCn + c]) * ow[c];
        s2 += bf2f(abuf[(size_t)p * FCn + c]) * aw[c];
    }
    pl[p]  = s1 + ob[0];
    sco[p] = s2 + ab[0];
}

// ---- MIL masked softmax + bag + attention maps ----------------------------

__device__ float blk_reduce(float v, float* sh, int domax) {
    int tid = threadIdx.x;
    sh[tid] = v; __syncthreads();
    for (int s = 128; s > 0; s >>= 1) {
        if (tid < s) sh[tid] = domax ? fmaxf(sh[tid], sh[tid + s]) : sh[tid] + sh[tid + s];
        __syncthreads();
    }
    float r = sh[0]; __syncthreads();
    return r;
}

__global__ void k_mil(const float* __restrict__ sco, const float* __restrict__ pl,
                      const int* __restrict__ zone, const int* __restrict__ cats,
                      float* __restrict__ maps, float* __restrict__ bagbuf) {
    __shared__ float sh[256];
    int b = blockIdx.x, tid = threadIdx.x;
    int cat = cats[b];
    float msc[16]; bool msk[16];
    float hasl = 0.f, maxl = -3.4e38f;
    for (int j = 0; j < 16; ++j) {
        int i = tid + j * 256;
        int z = zone[b * HW + i];
        bool mk = (z == cat) && (z > 0);
        float mval = mk ? sco[b * HW + i] : -1e9f;
        msk[j] = mk; msc[j] = mval;
        if (mk) hasl = 1.f;
        maxl = fmaxf(maxl, mval);
    }
    float maxv = blk_reduce(maxl, sh, 1);
    float hasv = blk_reduce(hasl, sh, 1);
    float se = 0.f;
    for (int j = 0; j < 16; ++j) se += __expf(msc[j] - maxv);
    float sumv = blk_reduce(se, sh, 0);
    float bagl = 0.f;
    for (int j = 0; j < 16; ++j) {
        int i = tid + j * 256;
        float w = __expf(msc[j] - maxv) / sumv;
        maps[b * HW + i] = (hasv > 0.f) ? w : 0.f;
        if (msk[j]) bagl += pl[b * HW + i] * w;
    }
    float bag = blk_reduce(bagl, sh, 0);
    if (tid == 0) { bagbuf[b * 2] = bag; bagbuf[b * 2 + 1] = hasv; }
}

__global__ void k_loss(const float* __restrict__ bagbuf, const float* __restrict__ labels,
                       float* __restrict__ out0) {
    if (threadIdx.x != 0 || blockIdx.x != 0) return;
    float loss = 0.f;
    for (int b = 0; b < Bn; ++b) {
        float x = (bagbuf[b * 2 + 1] > 0.f) ? bagbuf[b * 2] : 0.f;
        float sp = fmaxf(x, 0.f) + log1pf(__expf(-fabsf(x)));
        loss += sp - x * labels[b];
    }
    out0[0] = loss / (float)Bn;
}

// ---------------------------------------------------------------------------

extern "C" void kernel_launch(void* const* d_in, const int* in_sizes, int n_in,
                              void* d_out, int out_size, void* d_ws, size_t ws_size,
                              hipStream_t stream) {
    (void)in_sizes; (void)n_in; (void)out_size; (void)ws_size;
    const float* state = (const float*)d_in[0];
    const float* trig  = (const float*)d_in[1];
    const float* gnn   = (const float*)d_in[2];
    const int*   zone  = (const int*)d_in[3];
    const int*   cats  = (const int*)d_in[4];
    const float* labels = (const float*)d_in[5];
    const float* s2t_qkv_w = (const float*)d_in[6];
    const float* s2t_qkv_b = (const float*)d_in[7];
    const float* s2t_out_w = (const float*)d_in[8];
    const float* s2t_out_b = (const float*)d_in[9];
    const float* t2s_qkv_w = (const float*)d_in[10];
    const float* t2s_qkv_b = (const float*)d_in[11];
    const float* t2s_out_w = (const float*)d_in[12];
    const float* t2s_out_b = (const float*)d_in[13];
    const float* fconv1_w  = (const float*)d_in[14];
    const float* fbn1_g = (const float*)d_in[15];
    const float* fbn1_b = (const float*)d_in[16];
    const float* fconv2_w  = (const float*)d_in[17];
    const float* fbn2_g = (const float*)d_in[18];
    const float* fbn2_b = (const float*)d_in[19];
    const float* outc_w = (const float*)d_in[20];
    const float* outc_b = (const float*)d_in[21];
    const float* attn1_w = (const float*)d_in[22];
    const float* attn1_b = (const float*)d_in[23];
    const float* attn2_w = (const float*)d_in[24];
    const float* attn2_b = (const float*)d_in[25];
    float* dout = (float*)d_out;

    char* base = (char*)d_ws;
    size_t off = 0;
    auto alloc = [&](size_t bytes) -> void* {
        void* p = base + off;
        off = (off + bytes + 255) & ~(size_t)255;
        return p;
    };
    typedef unsigned short us;
    us* Xs = (us*)alloc((size_t)MTOT * E * 2);
    us* Xt = (us*)alloc((size_t)MTOT * E * 2);
    us* wS_qkv = (us*)alloc(3 * E * E * 2);
    us* wS_out = (us*)alloc(E * E * 2);
    us* wT_qkv = (us*)alloc(3 * E * E * 2);
    us* wT_out = (us*)alloc(E * E * 2);
    us* wA1    = (us*)alloc(E * E * 2);
    us* wc1    = (us*)alloc((size_t)FCn * 9 * CP1 * 2);
    us* wc2    = (us*)alloc((size_t)FCn * 9 * CP2 * 2);
    us* Qs = (us*)alloc((size_t)MTOT * E * 2);
    us* Ks = (us*)alloc((size_t)MTOT * E * 2);
    us* Vs = (us*)alloc((size_t)MTOT * E * 2);
    us* Qt = (us*)alloc((size_t)MTOT * E * 2);
    us* Kt = (us*)alloc((size_t)MTOT * E * 2);
    us* Vt = (us*)alloc((size_t)MTOT * E * 2);
    us* ctxS = (us*)alloc((size_t)MTOT * E * 2);
    us* ctxT = (us*)alloc((size_t)MTOT * E * 2);
    us* fused = (us*)alloc((size_t)MTOT * CP1 * 2);
    float* convout = (float*)alloc((size_t)MTOT * FCn * 4);
    us* feat1 = (us*)alloc((size_t)MTOT * FCn * 2);
    us* feat2 = (us*)alloc((size_t)MTOT * FCn * 2);
    us* abuf  = (us*)alloc((size_t)MTOT * FCn * 2);
    float* scaleB = (float*)alloc(FCn * 4);
    float* shiftB = (float*)alloc(FCn * 4);
    float* pl  = (float*)alloc(MTOT * 4);
    float* sco = (float*)alloc(MTOT * 4);
    float* bagbuf = (float*)alloc(4 * 4);

    // 1) convert activations + weights to bf16
    int totNLC = MTOT * E;
    k_nchw_nlc<<<(totNLC + 255) / 256, 256, 0, stream>>>(state, Xs, E, HW, totNLC);
    k_nchw_nlc<<<(totNLC + 255) / 256, 256, 0, stream>>>(trig,  Xt, E, HW, totNLC);
    k_cvt<<<(3 * E * E + 255) / 256, 256, 0, stream>>>(s2t_qkv_w, wS_qkv, 3 * E * E);
    k_cvt<<<(E * E + 255) / 256, 256, 0, stream>>>(s2t_out_w, wS_out, E * E);
    k_cvt<<<(3 * E * E + 255) / 256, 256, 0, stream>>>(t2s_qkv_w, wT_qkv, 3 * E * E);
    k_cvt<<<(E * E + 255) / 256, 256, 0, stream>>>(t2s_out_w, wT_out, E * E);
    k_cvt<<<(E * E + 255) / 256, 256, 0, stream>>>(attn1_w, wA1, E * E);
    k_convw<<<(FCn * 9 * CP1 + 255) / 256, 256, 0, stream>>>(fconv1_w, wc1, 257, CP1);
    k_convw<<<(FCn * 9 * CP2 + 255) / 256, 256, 0, stream>>>(fconv2_w, wc2, 128, CP2);

    // 2) QKV projections (WMMA GEMMs), M=8192 N=128 K=128
    int Mt = MTOT / 16, Nt = E / 16, Ksteps = E / 32;
    int gblk = (Mt * Nt * 32 + 127) / 128;
    k_gemm<<<gblk, 128, 0, stream>>>(Xs, E, wS_qkv,             E, s2t_qkv_b,       Qs, nullptr, E, 0, Mt, Nt, Ksteps, 0);
    k_gemm<<<gblk, 128, 0, stream>>>(Xt, E, wS_qkv + E * E,     E, s2t_qkv_b + E,   Ks, nullptr, E, 0, Mt, Nt, Ksteps, 0);
    k_gemm<<<gblk, 128, 0, stream>>>(Xt, E, wS_qkv + 2 * E * E, E, s2t_qkv_b + 2*E, Vs, nullptr, E, 0, Mt, Nt, Ksteps, 0);
    k_gemm<<<gblk, 128, 0, stream>>>(Xt, E, wT_qkv,             E, t2s_qkv_b,       Qt, nullptr, E, 0, Mt, Nt, Ksteps, 0);
    k_gemm<<<gblk, 128, 0, stream>>>(Xs, E, wT_qkv + E * E,     E, t2s_qkv_b + E,   Kt, nullptr, E, 0, Mt, Nt, Ksteps, 0);
    k_gemm<<<gblk, 128, 0, stream>>>(Xs, E, wT_qkv + 2 * E * E, E, t2s_qkv_b + 2*E, Vt, nullptr, E, 0, Mt, Nt, Ksteps, 0);

    // 3) flash attention (one wave / 16-query tile / head / batch)
    k_attn<<<Bn * NHEAD * (HW / 16), 32, 0, stream>>>(Qs, Ks, Vs, ctxS);
    k_attn<<<Bn * NHEAD * (HW / 16), 32, 0, stream>>>(Qt, Kt, Vt, ctxT);

    // 4) output projections straight into padded fused buffer
    k_gemm<<<gblk, 128, 0, stream>>>(ctxS, E, wS_out, E, s2t_out_b, fused, nullptr, CP1, 0,   Mt, Nt, Ksteps, 0);
    k_gemm<<<gblk, 128, 0, stream>>>(ctxT, E, wT_out, E, t2s_out_b, fused, nullptr, CP1, 128, Mt, Nt, Ksteps, 0);
    k_fused_tail<<<(MTOT * 32 + 255) / 256, 256, 0, stream>>>(gnn, fused);

    // 5) conv1 -> BN1 -> relu
    int cblk = ((MTOT / 16) * (FCn / 16) * 32 + 127) / 128;
    k_conv<<<cblk, 128, 0, stream>>>(fused, wc1, convout, CP1);
    k_bnstats<<<FCn, 256, 0, stream>>>(convout, fbn1_g, fbn1_b, scaleB, shiftB);
    k_bnapply<<<(MTOT * FCn + 255) / 256, 256, 0, stream>>>(convout, scaleB, shiftB, feat1);

    // 6) conv2 -> BN2 -> relu
    k_conv<<<cblk, 128, 0, stream>>>(feat1, wc2, convout, CP2);
    k_bnstats<<<FCn, 256, 0, stream>>>(convout, fbn2_g, fbn2_b, scaleB, shiftB);
    k_bnapply<<<(MTOT * FCn + 255) / 256, 256, 0, stream>>>(convout, scaleB, shiftB, feat2);

    // 7) attn1 1x1 conv (GEMM + bias + relu), then scalar heads
    k_gemm<<<gblk, 128, 0, stream>>>(feat2, FCn, wA1, FCn, attn1_b, abuf, nullptr, FCn, 0, Mt, Nt, Ksteps, 1);
    k_heads<<<(MTOT + 255) / 256, 256, 0, stream>>>(feat2, abuf, outc_w, outc_b, attn2_w, attn2_b, pl, sco);

    // 8) MIL masked softmax, attention maps, BCE loss
    k_mil<<<Bn, 256, 0, stream>>>(sco, pl, zone, cats, dout + 1, bagbuf);
    k_loss<<<1, 1, 0, stream>>>(bagbuf, labels, dout);
}
